// MulHeadTransformerLayer_35802847379558
// MI455X (gfx1250) — matrix operand order repaired and checked by reading, not compiled, not run on hardware
//
#include <hip/hip_runtime.h>
#include <hip/hip_bf16.h>

typedef _Float16 v16h __attribute__((ext_vector_type(16)));
typedef _Float16 v8h  __attribute__((ext_vector_type(8)));
typedef float    v8f  __attribute__((ext_vector_type(8)));

// ---- problem constants (match reference) ----
constexpr int Bb  = 8;
constexpr int Nn  = 2048;
constexpr int Kk  = 16;
constexpr int Hh  = 4;
constexpr int HDc = 64;
constexpr int DM  = 256;   // D_MODEL
constexpr int DP  = 128;   // D_PTS
constexpr int BN  = Bb * Nn;       // 16384 rows
constexpr int BNK = BN * Kk;       // 262144 neighbor rows
constexpr long RES_ELEMS = (long)BN * DP;  // res output elements

// ============================================================
// KNN: one thread per point, top-16 insertion sort in registers
// ============================================================
__global__ void knn_kernel(const float* __restrict__ xyz, int* __restrict__ knn) {
  int gid = blockIdx.x * blockDim.x + threadIdx.x;   // 0..BN-1
  int b = gid >> 11, n = gid & (Nn - 1);
  const float* base = xyz + (long)b * Nn * 3;
  float px = base[n*3+0], py = base[n*3+1], pz = base[n*3+2];
  float bd[Kk]; int bi[Kk];
#pragma unroll
  for (int j = 0; j < Kk; ++j) { bd[j] = 3.4e38f; bi[j] = 0; }
  for (int m = 0; m < Nn; ++m) {
    float dx = px - base[m*3+0];
    float dy = py - base[m*3+1];
    float dz = pz - base[m*3+2];
    float d = dx*dx + dy*dy + dz*dz;
    if (d < bd[Kk-1]) {
      int j = Kk - 1;
      while (j > 0 && bd[j-1] > d) { bd[j] = bd[j-1]; bi[j] = bi[j-1]; --j; }
      bd[j] = d; bi[j] = m;
    }
  }
#pragma unroll
  for (int j = 0; j < Kk; ++j) knn[(long)gid * Kk + j] = bi[j];
}

// ============================================================
// f32 -> f16 row-major convert
// ============================================================
__global__ void conv_f32_to_f16(const float* __restrict__ in, _Float16* __restrict__ out, int n) {
  for (int i = blockIdx.x * blockDim.x + threadIdx.x; i < n; i += gridDim.x * blockDim.x)
    out[i] = (_Float16)in[i];
}

// ============================================================
// Pack weight [KC,NC] f32 (row-major) into WMMA B-fragment-major f16.
// Fragment f covers K-tile (32) x N-tile (16). Within a fragment,
// lane l element e holds W[K = kt*32 + 16*(l>>4) + e, N = nt*16 + (l&15)],
// stored contiguously: out[f*512 + l*16 + e]  -> one 32B LDS read per lane.
// ============================================================
__global__ void pack_weight_frag(const float* __restrict__ W, _Float16* __restrict__ out,
                                 int KC, int NC) {
  int total = KC * NC;
  int nT = NC >> 4;
  for (int i = blockIdx.x * blockDim.x + threadIdx.x; i < total; i += gridDim.x * blockDim.x) {
    int f = i >> 9;
    int l = (i >> 4) & 31;
    int e = i & 15;
    int kt = f / nT, nt = f - kt * nT;
    int Kr   = kt * 32 + ((l >> 4) << 4) + e;
    int Ncol = nt * 16 + (l & 15);
    out[i] = (_Float16)W[Kr * NC + Ncol];
  }
}

// ============================================================
// WMMA GEMM: out[M,NC] = A[M,KC](f16) @ W[KC,NC] (+bias) (opt relu)
// Block = 256 threads = 8 waves, 128 rows/block; wave owns 16 rows.
// Full weight matrix staged in LDS (fragment-major), KC*NC*2 bytes.
// A fragment (16x32 f16): lane l, element e -> A[r0+(l&15), kb + 16*(l>>4) + e]
//   == two contiguous 8-half global loads per lane per k-step.
// C/D (16x16 f32): VGPR r -> row r + 8*(l>>4), col (l&15).
// ============================================================
template<int KC, int NC, bool BIAS, bool RELU, bool OUTF, bool OUTH>
__global__ void wmma_gemm_kernel(const _Float16* __restrict__ A,
                                 const _Float16* __restrict__ Wfrag,
                                 const float* __restrict__ bias,
                                 float* __restrict__ outF,
                                 _Float16* __restrict__ outH) {
  extern __shared__ char smem_raw[];
  _Float16* ldsW = (_Float16*)smem_raw;

  const int tid = threadIdx.x;
  constexpr int WT = KC * NC;
  for (int i = tid * 8; i < WT; i += 256 * 8)
    *(v8h*)(ldsW + i) = *(const v8h*)(Wfrag + i);
  __syncthreads();

  const int wave = tid >> 5;
  const int lane = tid & 31;
  const int hl   = lane >> 4;
  const int l15  = lane & 15;
  const long r0  = (long)blockIdx.x * 128 + wave * 16;
  const _Float16* arow = A + (r0 + l15) * KC;

  for (int ncb = 0; ncb < NC; ncb += 64) {
    v8f acc[4] = {};
    for (int kb = 0; kb < KC; kb += 32) {
      union { v16h v; v8h h[2]; } af;
      af.h[0] = *(const v8h*)(arow + kb +     16 * hl);
      af.h[1] = *(const v8h*)(arow + kb + 8 + 16 * hl);
      const int ktile = kb >> 5;
#pragma unroll
      for (int s = 0; s < 4; ++s) {
        const int frag = ktile * (NC >> 4) + (ncb >> 4) + s;
        const v16h bf = *(const v16h*)(ldsW + frag * 512 + lane * 16);
        acc[s] = __builtin_amdgcn_wmma_f32_16x16x32_f16(
            false, af.v, false, bf, (short)0, acc[s], false, false);
      }
    }
#pragma unroll
    for (int s = 0; s < 4; ++s) {
      const int col = ncb + s * 16 + l15;
      const float bv = BIAS ? bias[col] : 0.0f;
#pragma unroll
      for (int r = 0; r < 8; ++r) {
        const long row = r0 + r + 8 * hl;
        float vv = acc[s][r] + bv;
        if (RELU) vv = fmaxf(vv, 0.0f);
        if (OUTF) outF[row * NC + col] = vv;
        if (OUTH) outH[row * NC + col] = (_Float16)vv;
      }
    }
  }
}

// ============================================================
// pos hidden: hidden[r,c] = relu(b1[c] + rel·delta_w1[:,c]), K-dim=3 (VALU)
// one block per neighbor-row r = ((b*N+n)*K + kk), 256 threads = cols
// ============================================================
__global__ void pos_hidden_kernel(const float* __restrict__ xyz,
                                  const int* __restrict__ knn,
                                  const float* __restrict__ w1,   // [3,256]
                                  const float* __restrict__ b1,
                                  _Float16* __restrict__ hiddenH) {
  long r = blockIdx.x;
  int  c = threadIdx.x;
  long bn = r >> 4;
  int b = (int)(bn >> 11);
  int n = (int)(bn & (Nn - 1));
  int id = knn[r];
  const float* base = xyz + (long)b * Nn * 3;
  float rx = base[n*3+0] - base[id*3+0];
  float ry = base[n*3+1] - base[id*3+1];
  float rz = base[n*3+2] - base[id*3+2];
  float v = b1[c] + rx * w1[c] + ry * w1[DM + c] + rz * w1[2*DM + c];
  hiddenH[r * DM + c] = (_Float16)fmaxf(v, 0.0f);
}

// ============================================================
// a = q - gather(k) + pos  (f16 rows for gamma GEMM; c = h*64+d)
// ============================================================
__global__ void a_assemble_kernel(const float* __restrict__ qF,
                                  const float* __restrict__ kF,
                                  const _Float16* __restrict__ posH,
                                  const int* __restrict__ knn,
                                  _Float16* __restrict__ aH) {
  long r = blockIdx.x;
  int  c = threadIdx.x;
  long bn = r >> 4;
  int b = (int)(bn >> 11);
  int id = knn[r];
  float a = qF[bn * DM + c] - kF[((long)b * Nn + id) * DM + c] + (float)posH[r * DM + c];
  aH[r * DM + c] = (_Float16)a;
}

// ============================================================
// per-channel softmax over K=16 neighbors + weighted sum of (v+pos)
// block = (b,n); thread = channel c = h*64+d. Writes attn_out + res(f16).
// ============================================================
__global__ void attn_kernel(const float* __restrict__ a2,
                            const int* __restrict__ knn,
                            const float* __restrict__ vF,
                            const _Float16* __restrict__ posH,
                            _Float16* __restrict__ resH,
                            float* __restrict__ attnOut) {
  int bn = blockIdx.x;
  int b = bn >> 11, n = bn & (Nn - 1);
  int c = threadIdx.x;
  int h = c >> 6, d = c & 63;
  float vals[Kk];
#pragma unroll
  for (int kk = 0; kk < Kk; ++kk) {
    long rowA = ((long)bn * Kk + kk) * Hh + h;           // gamma row id
    vals[kk] = a2[rowA * HDc + d] * 0.125f;              // / sqrt(HD)
  }
  float m = vals[0];
#pragma unroll
  for (int kk = 1; kk < Kk; ++kk) m = fmaxf(m, vals[kk]);
  float s = 0.0f;
#pragma unroll
  for (int kk = 0; kk < Kk; ++kk) { vals[kk] = __expf(vals[kk] - m); s += vals[kk]; }
  float inv = 1.0f / s;
  float acc = 0.0f;
  long attnBase = (((long)b * Hh + h) * Nn + n) * (Kk * HDc);
#pragma unroll
  for (int kk = 0; kk < Kk; ++kk) {
    float p = vals[kk] * inv;
    attnOut[attnBase + kk * HDc + d] = p;
    int id = knn[(long)bn * Kk + kk];
    float vv = vF[((long)b * Nn + id) * DM + c];
    float pp = (float)posH[((long)bn * Kk + kk) * DM + c];
    acc += p * (vv + pp);
  }
  resH[(long)bn * DM + c] = (_Float16)acc;
}

// ============================================================
// LayerNorm over DIM, one wave32 per row (8 rows / 256-thread block)
// ============================================================
template<int DIM, bool OUTF, bool OUTH, bool ADDPRE>
__global__ void layernorm_kernel(const float* __restrict__ X,
                                 const float* __restrict__ gam,
                                 const float* __restrict__ bet,
                                 const float* __restrict__ pre,
                                 float* __restrict__ outF,
                                 _Float16* __restrict__ outH) {
  int wave = threadIdx.x >> 5, lane = threadIdx.x & 31;
  long row = (long)blockIdx.x * 8 + wave;
  const float* x = X + row * DIM;
  constexpr int PER = DIM / 32;
  float v[PER]; float s = 0.0f, sq = 0.0f;
#pragma unroll
  for (int i = 0; i < PER; ++i) {
    v[i] = x[lane + i * 32]; s += v[i]; sq += v[i] * v[i];
  }
#pragma unroll
  for (int m = 16; m > 0; m >>= 1) {
    s  += __shfl_xor(s,  m, 32);
    sq += __shfl_xor(sq, m, 32);
  }
  float mean = s * (1.0f / DIM);
  float var  = sq * (1.0f / DIM) - mean * mean;
  float inv  = rsqrtf(var + 1e-5f);
#pragma unroll
  for (int i = 0; i < PER; ++i) {
    int c = lane + i * 32;
    float y = (v[i] - mean) * inv * gam[c] + bet[c];
    if (ADDPRE) y += pre[row * DIM + c];
    if (OUTF) outF[row * DIM + c] = y;
    if (OUTH) outH[row * DIM + c] = (_Float16)y;
  }
}

// ============================================================
// launcher
// ============================================================
extern "C" void kernel_launch(void* const* d_in, const int* in_sizes, int n_in,
                              void* d_out, int out_size, void* d_ws, size_t ws_size,
                              hipStream_t stream) {
  (void)in_sizes; (void)n_in; (void)out_size; (void)ws_size;
  const float* xyz      = (const float*)d_in[0];
  const float* features = (const float*)d_in[1];
  const float* fc1_w    = (const float*)d_in[2];
  const float* fc1_b    = (const float*)d_in[3];
  const float* fc2_w    = (const float*)d_in[4];
  const float* fc2_b    = (const float*)d_in[5];
  const float* delta_w1 = (const float*)d_in[6];
  const float* delta_b1 = (const float*)d_in[7];
  const float* delta_w2 = (const float*)d_in[8];
  const float* delta_b2 = (const float*)d_in[9];
  const float* gamma_w1 = (const float*)d_in[10];
  const float* gamma_b1 = (const float*)d_in[11];
  const float* gamma_w2 = (const float*)d_in[12];
  const float* gamma_b2 = (const float*)d_in[13];
  const float* wq       = (const float*)d_in[14];
  const float* wk       = (const float*)d_in[15];
  const float* wv       = (const float*)d_in[16];
  const float* proj_w   = (const float*)d_in[17];
  const float* ln1_g    = (const float*)d_in[18];
  const float* ln1_b    = (const float*)d_in[19];
  const float* ln2_g    = (const float*)d_in[20];
  const float* ln2_b    = (const float*)d_in[21];

  char* ws = (char*)d_ws;
  size_t off = 0;
  auto alloc = [&](size_t bytes) -> void* {
    void* p = ws + off;
    off += (bytes + 255) & ~(size_t)255;
    return p;
  };
  int*      knn     = (int*)     alloc((size_t)BNK * 4);
  _Float16* feat_h  = (_Float16*)alloc((size_t)BN * DP * 2);
  _Float16* x_h     = (_Float16*)alloc((size_t)BN * DM * 2);
  float*    q_f     = (float*)   alloc((size_t)BN * DM * 4);
  float*    k_f     = (float*)   alloc((size_t)BN * DM * 4);
  float*    v_f     = (float*)   alloc((size_t)BN * DM * 4);
  _Float16* fc1_wh  = (_Float16*)alloc((size_t)DP * DM * 2);
  _Float16* wq_h    = (_Float16*)alloc((size_t)DM * DM * 2);
  _Float16* wk_h    = (_Float16*)alloc((size_t)DM * DM * 2);
  _Float16* wv_h    = (_Float16*)alloc((size_t)DM * DM * 2);
  _Float16* dw2_h   = (_Float16*)alloc((size_t)DM * DM * 2);
  _Float16* g1w_h   = (_Float16*)alloc((size_t)HDc * HDc * 2);
  _Float16* g2w_h   = (_Float16*)alloc((size_t)HDc * HDc * 2);
  _Float16* proj_h  = (_Float16*)alloc((size_t)DM * DM * 2);
  _Float16* fc2_wh  = (_Float16*)alloc((size_t)DM * DP * 2);
  _Float16* hidden_h= (_Float16*)alloc((size_t)BNK * DM * 2);  // reused as a_h
  _Float16* pos_h   = (_Float16*)alloc((size_t)BNK * DM * 2);
  _Float16* g1_h    = (_Float16*)alloc((size_t)BNK * DM * 2);
  float*    a2_f    = (float*)   alloc((size_t)BNK * DM * 4);
  _Float16* res_h   = (_Float16*)alloc((size_t)BN * DM * 2);
  float*    t1_f    = (float*)   alloc((size_t)BN * DM * 4);
  _Float16* y1_h    = (_Float16*)alloc((size_t)BN * DM * 2);
  float*    t2_f    = (float*)   alloc((size_t)BN * DP * 4);

  float* out_res  = (float*)d_out;
  float* out_attn = out_res + RES_ELEMS;

  // --- KNN + precision conversion / weight fragment packing ---
  knn_kernel<<<BN / 256, 256, 0, stream>>>(xyz, knn);
  conv_f32_to_f16<<<256, 256, 0, stream>>>(features, feat_h, BN * DP);
  pack_weight_frag<<<64, 256, 0, stream>>>(fc1_w,    fc1_wh, DP,  DM);
  pack_weight_frag<<<64, 256, 0, stream>>>(wq,       wq_h,   DM,  DM);
  pack_weight_frag<<<64, 256, 0, stream>>>(wk,       wk_h,   DM,  DM);
  pack_weight_frag<<<64, 256, 0, stream>>>(wv,       wv_h,   DM,  DM);
  pack_weight_frag<<<64, 256, 0, stream>>>(delta_w2, dw2_h,  DM,  DM);
  pack_weight_frag<<<64, 256, 0, stream>>>(gamma_w1, g1w_h,  HDc, HDc);
  pack_weight_frag<<<64, 256, 0, stream>>>(gamma_w2, g2w_h,  HDc, HDc);
  pack_weight_frag<<<64, 256, 0, stream>>>(proj_w,   proj_h, DM,  DM);
  pack_weight_frag<<<64, 256, 0, stream>>>(fc2_w,    fc2_wh, DM,  DP);

  // --- fc1: x = features @ fc1_w + fc1_b  -> f16 ---
  wmma_gemm_kernel<DP, DM, true, false, false, true>
      <<<BN / 128, 256, DP * DM * 2, stream>>>(feat_h, fc1_wh, fc1_b, nullptr, x_h);

  // --- q/k/v GEMMs -> f32 ---
  wmma_gemm_kernel<DM, DM, false, false, true, false>
      <<<BN / 128, 256, DM * DM * 2, stream>>>(x_h, wq_h, nullptr, q_f, nullptr);
  wmma_gemm_kernel<DM, DM, false, false, true, false>
      <<<BN / 128, 256, DM * DM * 2, stream>>>(x_h, wk_h, nullptr, k_f, nullptr);
  wmma_gemm_kernel<DM, DM, false, false, true, false>
      <<<BN / 128, 256, DM * DM * 2, stream>>>(x_h, wv_h, nullptr, v_f, nullptr);

  // --- pos MLP: relu(rel@w1+b1) (VALU) then @w2+b2 (WMMA) ---
  pos_hidden_kernel<<<BNK, 256, 0, stream>>>(xyz, knn, delta_w1, delta_b1, hidden_h);
  wmma_gemm_kernel<DM, DM, true, false, false, true>
      <<<BNK / 128, 256, DM * DM * 2, stream>>>(hidden_h, dw2_h, delta_b2, nullptr, pos_h);

  // --- a = q - gather(k) + pos  (reuse hidden buffer as a) ---
  a_assemble_kernel<<<BNK, 256, 0, stream>>>(q_f, k_f, pos_h, knn, hidden_h);

  // --- gamma MLP over (B,N,K,H) rows of 64: relu(a@g1+b1)@g2+b2 ---
  wmma_gemm_kernel<HDc, HDc, true, true, false, true>
      <<<(BNK * Hh) / 128, 256, HDc * HDc * 2, stream>>>(hidden_h, g1w_h, gamma_b1, nullptr, g1_h);
  wmma_gemm_kernel<HDc, HDc, true, false, true, false>
      <<<(BNK * Hh) / 128, 256, HDc * HDc * 2, stream>>>(g1_h, g2w_h, gamma_b2, a2_f, nullptr);

  // --- softmax over K + weighted sum of (v+pos); writes attn_out ---
  attn_kernel<<<BN, 256, 0, stream>>>(a2_f, knn, v_f, pos_h, res_h, out_attn);

  // --- proj GEMM -> LN1 -> fc2 GEMM -> LN2 + residual ---
  wmma_gemm_kernel<DM, DM, false, false, true, false>
      <<<BN / 128, 256, DM * DM * 2, stream>>>(res_h, proj_h, nullptr, t1_f, nullptr);
  layernorm_kernel<DM, false, true, false>
      <<<BN / 8, 256, 0, stream>>>(t1_f, ln1_g, ln1_b, nullptr, nullptr, y1_h);
  wmma_gemm_kernel<DM, DP, true, false, true, false>
      <<<BN / 128, 256, DM * DP * 2, stream>>>(y1_h, fc2_wh, fc2_b, t2_f, nullptr);
  layernorm_kernel<DP, true, false, true>
      <<<BN / 8, 256, 0, stream>>>(t2_f, ln2_g, ln2_b, features, out_res, nullptr);
}
